// GraphSAGE_12223476924781
// MI455X (gfx1250) — matrix operand order repaired and checked by reading, not compiled, not run on hardware
//
#include <hip/hip_runtime.h>
#include <hip/hip_bf16.h>

typedef __attribute__((ext_vector_type(16))) _Float16 v16h;
typedef __attribute__((ext_vector_type(8)))  _Float16 v8h;
typedef __attribute__((ext_vector_type(4)))  _Float16 v4h;
typedef __attribute__((ext_vector_type(2)))  _Float16 v2h;
typedef __attribute__((ext_vector_type(8)))  float    v8f;

// ---------------------------------------------------------------------------
// WMMA helpers (CDNA5 / gfx1250, wave32)
// ---------------------------------------------------------------------------

__device__ __forceinline__ v8f wmma_f16(v16h a, v16h b, v8f c) {
    // D = A(16x32 f16) * B(32x16 f16) + C(16x16 f32)
    return __builtin_amdgcn_wmma_f32_16x16x32_f16(
        /*neg_a=*/false, a, /*neg_b=*/false, b,
        /*c_mod=*/(short)0, c, /*reuse_a=*/false, /*reuse_b=*/false);
}

__device__ __forceinline__ v16h cat8(v8h lo, v8h hi) {
    return __builtin_shufflevector(lo, hi, 0, 1, 2, 3, 4, 5, 6, 7,
                                           8, 9, 10, 11, 12, 13, 14, 15);
}

// A fragment from row-major f16 [nrows x 128].
// Lane group g: halves 0..7 <- K = kbase+8g.. ; halves 8..15 <- K = kbase+16+8g..
__device__ __forceinline__ v16h load_a_frag(const _Float16* __restrict__ X,
                                            int row, int kbase, int g) {
    const _Float16* p = X + (size_t)row * 128 + kbase + g * 8;
    v8h lo = *(const v8h*)(p);        // 16B aligned
    v8h hi = *(const v8h*)(p + 16);
    return cat8(lo, hi);
}

// B fragment from pre-transposed f16 weights WT[n][k] (n-major, ld = 128).
// Lane group g: halves 0..7 <- K = kbase+8g.. ; halves 8..15 <- K = kbase+16+8g..
__device__ __forceinline__ v16h load_b_frag(const _Float16* __restrict__ WT,
                                            int n, int kbase, int g) {
    const _Float16* p = WT + (size_t)n * 128 + kbase + g * 8;
    v8h lo = *(const v8h*)(p);        // 16B aligned
    v8h hi = *(const v8h*)(p + 16);
    return cat8(lo, hi);
}

// CDNA5 packed-f16 atomic add (L2 RMW, no return) — one op adds 2 halfs.
__device__ __forceinline__ void atomic_pk_add_f16(_Float16* addr, v2h val) {
    asm volatile("global_atomic_pk_add_f16 %0, %1, off"
                 :: "v"(addr), "v"(val)
                 : "memory");
}

// ---------------------------------------------------------------------------
// Plain GEMM: Out_f16[nrows x NC] = X_f16[nrows x 128] @ WT_f16[NC x 128]^T
// Block = 256 threads = 8 waves; each wave owns 16 rows.
// FULL=true: wave's 16 rows all in range -> no clamps, no store guards.
// ---------------------------------------------------------------------------
template <int NC, bool FULL>
__device__ __forceinline__ void gemm_body(
    const _Float16* __restrict__ X, const _Float16* __restrict__ WT,
    _Float16* __restrict__ Out, int nrows, int r0, int g, int ml) {
    int rowA = r0 + ml;
    if (!FULL) rowA = rowA < nrows ? rowA : nrows - 1;
    v16h a0 = load_a_frag(X, rowA, 0,  g);
    v16h a1 = load_a_frag(X, rowA, 32, g);
    v16h a2 = load_a_frag(X, rowA, 64, g);
    v16h a3 = load_a_frag(X, rowA, 96, g);

#pragma unroll
    for (int nt = 0; nt < NC / 16; ++nt) {
        const int n = nt * 16 + ml;
        v8f c = {};
        c = wmma_f16(a0, load_b_frag(WT, n, 0,  g), c);
        c = wmma_f16(a1, load_b_frag(WT, n, 32, g), c);
        c = wmma_f16(a2, load_b_frag(WT, n, 64, g), c);
        c = wmma_f16(a3, load_b_frag(WT, n, 96, g), c);
#pragma unroll
        for (int r = 0; r < 8; ++r) {
            const int row = r0 + r + 8 * g;
            if (FULL || row < nrows) Out[(size_t)row * NC + n] = (_Float16)c[r];
        }
    }
}

template <int NC>
__global__ __launch_bounds__(256) void gemm_wmma_store(
    const _Float16* __restrict__ X, const _Float16* __restrict__ WT,
    _Float16* __restrict__ Out, int nrows) {
    const int lane = threadIdx.x & 31;
    const int g = lane >> 4, ml = lane & 15;
    const int r0 = (blockIdx.x * 8 + (threadIdx.x >> 5)) * 16;
    if (r0 >= nrows) return;                       // uniform per wave
    if (r0 + 16 <= nrows) gemm_body<NC, true >(X, WT, Out, nrows, r0, g, ml);
    else                  gemm_body<NC, false>(X, WT, Out, nrows, r0, g, ml);
}

// ---------------------------------------------------------------------------
// Fused layer-1 tail: H_f16 = relu(AGGh/max(cnt,1) + X@W1r + b1)   (NC = 128)
// ---------------------------------------------------------------------------
template <bool FULL>
__device__ __forceinline__ void fused1_body(
    const _Float16* __restrict__ X, const _Float16* __restrict__ WT,
    const float* __restrict__ bias, const _Float16* __restrict__ AGG,
    const float* __restrict__ CNT, _Float16* __restrict__ H,
    int nrows, int r0, int g, int ml) {
    int rowA = r0 + ml;
    if (!FULL) rowA = rowA < nrows ? rowA : nrows - 1;
    v16h a0 = load_a_frag(X, rowA, 0,  g);
    v16h a1 = load_a_frag(X, rowA, 32, g);
    v16h a2 = load_a_frag(X, rowA, 64, g);
    v16h a3 = load_a_frag(X, rowA, 96, g);

    // Per-row mean reciprocals, hoisted out of the tile loops (8 loads, 8 rcp)
    float inv[8];
#pragma unroll
    for (int r = 0; r < 8; ++r) {
        int row = r0 + r + 8 * g;
        if (!FULL) row = row < nrows ? row : nrows - 1;
        inv[r] = __builtin_amdgcn_rcpf(fmaxf(CNT[row], 1.0f));
    }

#pragma unroll
    for (int nt = 0; nt < 8; ++nt) {
        const int n = nt * 16 + ml;
        v8f c = {};
        c = wmma_f16(a0, load_b_frag(WT, n, 0,  g), c);
        c = wmma_f16(a1, load_b_frag(WT, n, 32, g), c);
        c = wmma_f16(a2, load_b_frag(WT, n, 64, g), c);
        c = wmma_f16(a3, load_b_frag(WT, n, 96, g), c);
        const float bn = bias[n];
#pragma unroll
        for (int r = 0; r < 8; ++r) {
            const int row = r0 + r + 8 * g;
            if (FULL || row < nrows) {
                const size_t idx = (size_t)row * 128 + n;
                const float v = (float)AGG[idx] * inv[r] + c[r] + bn;
                H[idx] = (_Float16)fmaxf(v, 0.0f);
            }
        }
    }
}

__global__ __launch_bounds__(256) void fused_layer1(
    const _Float16* __restrict__ X, const _Float16* __restrict__ WT,
    const float* __restrict__ bias, const _Float16* __restrict__ AGG,
    const float* __restrict__ CNT, _Float16* __restrict__ H, int nrows) {
    const int lane = threadIdx.x & 31;
    const int g = lane >> 4, ml = lane & 15;
    const int r0 = (blockIdx.x * 8 + (threadIdx.x >> 5)) * 16;
    if (r0 >= nrows) return;
    if (r0 + 16 <= nrows) fused1_body<true >(X, WT, bias, AGG, CNT, H, nrows, r0, g, ml);
    else                  fused1_body<false>(X, WT, bias, AGG, CNT, H, nrows, r0, g, ml);
}

// ---------------------------------------------------------------------------
// Fused layer-2 tail: out = softmax(AGG2h/max(cnt,1) + H@W2r + b2)  (NC = 64)
// ---------------------------------------------------------------------------
template <bool FULL>
__device__ __forceinline__ void fused2_body(
    const _Float16* __restrict__ H, const _Float16* __restrict__ WT,
    const float* __restrict__ bias, const _Float16* __restrict__ AGG2,
    const float* __restrict__ CNT2, float* __restrict__ out,
    int nrows, int r0, int g, int ml) {
    int rowA = r0 + ml;
    if (!FULL) rowA = rowA < nrows ? rowA : nrows - 1;
    v16h a0 = load_a_frag(H, rowA, 0,  g);
    v16h a1 = load_a_frag(H, rowA, 32, g);
    v16h a2 = load_a_frag(H, rowA, 64, g);
    v16h a3 = load_a_frag(H, rowA, 96, g);

    v8f c[4] = {v8f{}, v8f{}, v8f{}, v8f{}};
#pragma unroll
    for (int nt = 0; nt < 4; ++nt) {
        const int n = nt * 16 + ml;
        c[nt] = wmma_f16(a0, load_b_frag(WT, n, 0,  g), c[nt]);
        c[nt] = wmma_f16(a1, load_b_frag(WT, n, 32, g), c[nt]);
        c[nt] = wmma_f16(a2, load_b_frag(WT, n, 64, g), c[nt]);
        c[nt] = wmma_f16(a3, load_b_frag(WT, n, 96, g), c[nt]);
    }

    const float b0 = bias[ml], b1 = bias[16 + ml], b2 = bias[32 + ml], b3 = bias[48 + ml];
#pragma unroll
    for (int r = 0; r < 8; ++r) {
        const int row = r0 + r + 8 * g;
        int rc = row;
        if (!FULL) rc = rc < nrows ? rc : nrows - 1;
        const float inv = __builtin_amdgcn_rcpf(fmaxf(CNT2[rc], 1.0f));
        const size_t base = (size_t)rc * 64 + ml;
        float z0 = (float)AGG2[base +  0] * inv + c[0][r] + b0;
        float z1 = (float)AGG2[base + 16] * inv + c[1][r] + b1;
        float z2 = (float)AGG2[base + 32] * inv + c[2][r] + b2;
        float z3 = (float)AGG2[base + 48] * inv + c[3][r] + b3;
        // row-wise softmax over 64 cols: 4 per lane x 16 lanes in this group
        float m = fmaxf(fmaxf(z0, z1), fmaxf(z2, z3));
        m = fmaxf(m, __shfl_xor(m, 8, 32));
        m = fmaxf(m, __shfl_xor(m, 4, 32));
        m = fmaxf(m, __shfl_xor(m, 2, 32));
        m = fmaxf(m, __shfl_xor(m, 1, 32));
        float e0 = __expf(z0 - m), e1 = __expf(z1 - m);
        float e2 = __expf(z2 - m), e3 = __expf(z3 - m);
        float s = e0 + e1 + e2 + e3;
        s += __shfl_xor(s, 8, 32);
        s += __shfl_xor(s, 4, 32);
        s += __shfl_xor(s, 2, 32);
        s += __shfl_xor(s, 1, 32);
        const float rs = __builtin_amdgcn_rcpf(s);
        if (FULL || row < nrows) {
            float* o = out + (size_t)row * 64 + ml;
            o[0] = e0 * rs; o[16] = e1 * rs; o[32] = e2 * rs; o[48] = e3 * rs;
        }
    }
}

__global__ __launch_bounds__(256) void fused_layer2(
    const _Float16* __restrict__ H, const _Float16* __restrict__ WT,
    const float* __restrict__ bias, const _Float16* __restrict__ AGG2,
    const float* __restrict__ CNT2, float* __restrict__ out, int nrows) {
    const int lane = threadIdx.x & 31;
    const int g = lane >> 4, ml = lane & 15;
    const int r0 = (blockIdx.x * 8 + (threadIdx.x >> 5)) * 16;
    if (r0 >= nrows) return;
    if (r0 + 16 <= nrows) fused2_body<true >(H, WT, bias, AGG2, CNT2, out, nrows, r0, g, ml);
    else                  fused2_body<false>(H, WT, bias, AGG2, CNT2, out, nrows, r0, g, ml);
}

// ---------------------------------------------------------------------------
// Edge scatter: aggh[dst] += feat_f16[src] (WID halfs), cnt[dst] += 1
// LPE lanes per edge; each lane: one b64 gather + two packed-f16 atomics.
// ---------------------------------------------------------------------------
template <int WID, int LPE>
__global__ __launch_bounds__(256) void scatter_kernel(
    const _Float16* __restrict__ feat, const int* __restrict__ src,
    const int* __restrict__ dst, _Float16* __restrict__ agg,
    float* __restrict__ cnt, int ne) {
    const size_t t = (size_t)blockIdx.x * blockDim.x + threadIdx.x;
    const size_t e = t / LPE;
    const int sl = (int)(t % LPE);
    if (e >= (size_t)ne) return;
    const int s = src[e];
    const int d = dst[e];
    const v4h v = *(const v4h*)(feat + (size_t)s * WID + sl * 4);
    const v2h lo = __builtin_shufflevector(v, v, 0, 1);
    const v2h hi = __builtin_shufflevector(v, v, 2, 3);
    _Float16* a = agg + (size_t)d * WID + sl * 4;   // 8B aligned
    atomic_pk_add_f16(a, lo);
    atomic_pk_add_f16(a + 2, hi);
    if (sl == 0) unsafeAtomicAdd(cnt + d, 1.0f);
}

// ---------------------------------------------------------------------------
// Prep kernels
// ---------------------------------------------------------------------------
__global__ __launch_bounds__(256) void zero_f4(float4* __restrict__ p, size_t n4) {
    const size_t i = (size_t)blockIdx.x * blockDim.x + threadIdx.x;
    if (i < n4) p[i] = make_float4(0.f, 0.f, 0.f, 0.f);
}

// out[i*8 .. i*8+7] = (f16) in[i*8 .. i*8+7]
__global__ __launch_bounds__(256) void f32_to_f16_vec(const float* __restrict__ in,
                                                      _Float16* __restrict__ out,
                                                      size_t n8) {
    const size_t i = (size_t)blockIdx.x * blockDim.x + threadIdx.x;
    if (i >= n8) return;
    const float4* p = (const float4*)in + i * 2;
    const float4 a = p[0], b = p[1];
    v8h o;
    o[0] = (_Float16)a.x; o[1] = (_Float16)a.y; o[2] = (_Float16)a.z; o[3] = (_Float16)a.w;
    o[4] = (_Float16)b.x; o[5] = (_Float16)b.y; o[6] = (_Float16)b.z; o[7] = (_Float16)b.w;
    *((v8h*)out + i) = o;
}

// WT[n*128 + k] = (f16) W[k*NC + n]   (W is [128 x NC] row-major)
__global__ __launch_bounds__(256) void weight_transpose_f16(
    const float* __restrict__ W, _Float16* __restrict__ WT, int NC) {
    const int t = blockIdx.x * blockDim.x + threadIdx.x;
    if (t >= NC * 128) return;
    const int n = t >> 7;       // t / 128
    const int k = t & 127;      // t % 128
    WT[t] = (_Float16)W[(size_t)k * NC + n];
}

// ---------------------------------------------------------------------------
// Host-side launch
// ---------------------------------------------------------------------------
extern "C" void kernel_launch(void* const* d_in, const int* in_sizes, int n_in,
                              void* d_out, int out_size, void* d_ws, size_t ws_size,
                              hipStream_t stream) {
    const float* x    = (const float*)d_in[0];   // [N,128]
    const float* W1l  = (const float*)d_in[1];   // [128,128]
    const float* W1r  = (const float*)d_in[2];   // [128,128]
    const float* b1   = (const float*)d_in[3];   // [128]
    const float* W2l  = (const float*)d_in[4];   // [128,64]
    const float* W2r  = (const float*)d_in[5];   // [128,64]
    const float* b2   = (const float*)d_in[6];   // [64]
    const int*   ei1  = (const int*)d_in[7];     // [2,E1]
    const int*   ei2  = (const int*)d_in[8];     // [2,E2]

    const int N  = in_sizes[0] / 128;
    const int E1 = in_sizes[7] / 2;
    const int E2 = in_sizes[8] / 2;
    float* out = (float*)d_out;                  // [N,64]

    // Workspace layout:
    //   AGG1h : N*128 f16 scatter accumulator (AGG2h aliases first N*64 halfs)
    //   CNT1  : N f32, CNT2 : N f32
    //   XH    : N*128 f16   HH : N*128 f16   PH : N*128 f16 (P then R)
    //   WT    : f16 weights W1l_t[128*128] W1r_t[128*128] W2l_t[64*128] W2r_t[64*128]
    _Float16* AGG1h = (_Float16*)d_ws;
    _Float16* AGG2h = AGG1h;                          // alias (layer-2 phase)
    float* CNT1 = (float*)(AGG1h + (size_t)N * 128);
    float* CNT2 = CNT1 + (size_t)N;
    _Float16* XH  = (_Float16*)(CNT2 + (size_t)N);
    _Float16* HH  = XH + (size_t)N * 128;
    _Float16* PH  = HH + (size_t)N * 128;
    _Float16* W1lT = PH + (size_t)N * 128;
    _Float16* W1rT = W1lT + 128 * 128;
    _Float16* W2lT = W1rT + 128 * 128;
    _Float16* W2rT = W2lT + 64 * 128;

    const int gemmBlocks = (N + 127) / 128;           // 8 waves x 16 rows / block

    // 1) zero AGG1h + CNT1 + CNT2 (N*64 + 2N floats worth, contiguous)
    {
        const size_t n4 = ((size_t)N * 66) / 4;
        zero_f4<<<(unsigned)((n4 + 255) / 256), 256, 0, stream>>>((float4*)AGG1h, n4);
    }
    // 2) one-time f16 conversions: x and all four weight matrices (transposed)
    {
        const size_t n8 = ((size_t)N * 128) / 8;
        f32_to_f16_vec<<<(unsigned)((n8 + 255) / 256), 256, 0, stream>>>(x, XH, n8);
        weight_transpose_f16<<<(128 * 128 + 255) / 256, 256, 0, stream>>>(W1l, W1lT, 128);
        weight_transpose_f16<<<(128 * 128 + 255) / 256, 256, 0, stream>>>(W1r, W1rT, 128);
        weight_transpose_f16<<<(64 * 128 + 255) / 256, 256, 0, stream>>>(W2l, W2lT, 64);
        weight_transpose_f16<<<(64 * 128 + 255) / 256, 256, 0, stream>>>(W2r, W2rT, 64);
    }
    // 3) P = x @ W1_l   (f16 out)
    gemm_wmma_store<128><<<gemmBlocks, 256, 0, stream>>>(XH, W1lT, PH, N);
    // 4) agg1[dst] += P[src]; cnt1[dst] += 1   (packed-f16 atomics)
    {
        const size_t total = (size_t)E1 * 32;
        scatter_kernel<128, 32><<<(unsigned)((total + 255) / 256), 256, 0, stream>>>(
            PH, ei1, ei1 + E1, AGG1h, CNT1, E1);
    }
    // 5) h = relu(agg1/cnt1 + x @ W1_r + b1)  -> HH (f16)
    fused_layer1<<<gemmBlocks, 256, 0, stream>>>(XH, W1rT, b1, AGG1h, CNT1, HH, N);
    // 6) re-zero AGG2h region (aliases AGG1h, safe: consumed by step 5)
    {
        const size_t n4 = ((size_t)N * 32) / 4;
        zero_f4<<<(unsigned)((n4 + 255) / 256), 256, 0, stream>>>((float4*)AGG2h, n4);
    }
    // 7) R = h @ W2_l   (f16 out, reuses PH)
    gemm_wmma_store<64><<<gemmBlocks, 256, 0, stream>>>(HH, W2lT, PH, N);
    // 8) agg2[dst] += R[src]; cnt2[dst] += 1
    {
        const size_t total = (size_t)E2 * 16;
        scatter_kernel<64, 16><<<(unsigned)((total + 255) / 256), 256, 0, stream>>>(
            PH, ei2, ei2 + E2, AGG2h, CNT2, E2);
    }
    // 9) out = softmax(agg2/cnt2 + h @ W2_r + b2)
    fused_layer2<<<gemmBlocks, 256, 0, stream>>>(HH, W2rT, b2, AGG2h, CNT2, out, N);
}
